// RNNModel_66881230733541
// MI455X (gfx1250) — matrix operand order repaired and checked by reading, not compiled, not run on hardware
//
#include <hip/hip_runtime.h>
#include <hip/hip_bf16.h>

// ---------------------------------------------------------------------------
// MI455X (gfx1250) implementation of the 2-layer LSTM+MHA RNN LM.
// Heavy GEMMs on v_wmma_f32_16x16x32_bf16 (wave32 WMMA), fp32 accumulate,
// with double-buffered GLOBAL_LOAD_ASYNC_TO_LDS staging (ASYNCcnt pipeline).
// Sequence: embed->bf16, weights->bf16, per layer { Gx = X*Wih^T + b (big
// GEMM), 64x { gates = Gx[t] + h*Whh^T (block-diag GEMM), cell+MHA+LN } },
// decoder GEMM (dominant 268 GFLOP) into d_out.
// ---------------------------------------------------------------------------

typedef __bf16 bf16_t;
typedef bf16_t v8bf  __attribute__((ext_vector_type(8)));
typedef bf16_t v16bf __attribute__((ext_vector_type(16)));
typedef float  v8f   __attribute__((ext_vector_type(8)));
typedef int    v4i   __attribute__((ext_vector_type(4)));

#define NLAYERS 2
#define NB      4
#define BS      256
#define NHID    1024
#define NINP    1024
#define NTOK    32000
#define T_SEQ   64
#define BATCH   64
#define DK      16

#if defined(__HIP_DEVICE_COMPILE__) && \
    __has_builtin(__builtin_amdgcn_global_load_async_to_lds_b128) && \
    __has_builtin(__builtin_amdgcn_s_wait_asynccnt)
#define HAVE_ASYNC_LDS 1
#else
#define HAVE_ASYNC_LDS 0
#endif

#if HAVE_ASYNC_LDS
// builtin signature (from hipcc diagnostic): (int4 AS1*, int4 AS3*, Imm, Imm)
typedef __attribute__((address_space(1))) v4i* gv4i_p;
typedef __attribute__((address_space(3))) v4i* lv4i_p;
#define GPTR(p) ((gv4i_p)(p))
#define LPTR(p) ((lv4i_p)(p))
#endif

// ---------------------------------------------------------------------------
// f32 -> bf16 conversion
// ---------------------------------------------------------------------------
__global__ void k_f32_to_bf16(const float* __restrict__ in,
                              bf16_t* __restrict__ out, long long n) {
  long long i = (long long)blockIdx.x * blockDim.x + threadIdx.x;
  if (i < n) out[i] = (bf16_t)in[i];
}

// ---------------------------------------------------------------------------
// Embedding gather -> bf16 activations  [T*B, NINP]
// ---------------------------------------------------------------------------
__global__ void k_embed(const int* __restrict__ tokens,
                        const float* __restrict__ embW,
                        bf16_t* __restrict__ x, long long total_elems) {
  long long i = (long long)blockIdx.x * blockDim.x + threadIdx.x;
  if (i >= total_elems) return;
  int row = (int)(i >> 10);          // T*B index
  int col = (int)(i & 1023);
  int tok = tokens[row];
  x[i] = (bf16_t)embW[(long long)tok * NINP + col];
}

// ---------------------------------------------------------------------------
// Batched WMMA GEMM:  C[M,N] = A[M,K](bf16) * W[N,K]^T(bf16)
//                               (+ bias0[n] + bias1[n]) (+ Cin[m,n])
// 256 threads = 8 waves.  Workgroup tile (WM*64) M x 128 N; wave w owns
// M-strip (w&3)*WM*16 (WM fragments of 16 rows) and N-strip (w>>2)*64
// (4 fragments of 16 cols) -> WM x 4 v8f accumulators, 4*WM WMMAs / kk step.
// K staged 64 wide in LDS, double-buffered via GLOBAL_LOAD_ASYNC_TO_LDS
// (pipeline: issue tile k+1, s_wait_asynccnt <= next-tile loads, barrier,
// compute tile k).  All problem dims divide the tiles exactly.
// ---------------------------------------------------------------------------
template <int WM>
__global__ void __launch_bounds__(256)
k_gemm_bf16(const bf16_t* __restrict__ A, int lda, long long bsA,
            const bf16_t* __restrict__ W, int ldw, long long bsW,
            float* __restrict__ C, int ldc, long long bsC,
            const float* __restrict__ Cin, int ldcin, long long bsCin,
            const float* __restrict__ bias0, const float* __restrict__ bias1,
            int K) {
  constexpr int MT = WM * 64;                 // workgroup M tile
  __shared__ bf16_t sA[2][MT * 64];           // [m][k], K-contiguous
  __shared__ bf16_t sB[2][128 * 64];          // [n][k], K-contiguous

  const int tid  = threadIdx.x;
  const int lane = tid & 31;
  const int wave = tid >> 5;
  const int msub = wave & 3;
  const int nsub = wave >> 2;
  const int lg   = lane >> 4;                 // lane group (0/1)
  const int l16  = lane & 15;

  const long long bz = blockIdx.z;
  const bf16_t* Ab = A + bz * bsA + (long long)(blockIdx.y * MT) * lda;
  const bf16_t* Wb = W + bz * bsW + (long long)(blockIdx.x * 128) * ldw;

  v8f acc[WM][4];
#pragma unroll
  for (int w = 0; w < WM; ++w)
#pragma unroll
    for (int j = 0; j < 4; ++j) acc[w][j] = (v8f)(0.0f);

#if HAVE_ASYNC_LDS
  // per-thread async loads per tile: 2*WM (A) + 4 (B)
  auto stage = [&](int buf, int k0) {
#pragma unroll
    for (int u = 0; u < 2 * WM; ++u) {
      int i = tid + u * 256;
      int r = i >> 3, cc = i & 7;
      __builtin_amdgcn_global_load_async_to_lds_b128(
          GPTR(&Ab[(long long)r * lda + k0 + cc * 8]),
          LPTR(&sA[buf][r * 64 + cc * 8]), 0, 0);
    }
#pragma unroll
    for (int u = 0; u < 4; ++u) {
      int i = tid + u * 256;
      int r = i >> 3, cc = i & 7;
      __builtin_amdgcn_global_load_async_to_lds_b128(
          GPTR(&Wb[(long long)r * ldw + k0 + cc * 8]),
          LPTR(&sB[buf][r * 64 + cc * 8]), 0, 0);
    }
  };
#endif

  const int nk = K >> 6;
#if HAVE_ASYNC_LDS
  stage(0, 0);
#endif

  for (int kt = 0; kt < nk; ++kt) {
    const int cur = kt & 1;
    const int k0  = kt << 6;
    (void)k0;

#if HAVE_ASYNC_LDS
    if (kt + 1 < nk) {
      stage(cur ^ 1, (kt + 1) << 6);
      // next tile leaves 2*WM+4 loads in flight; in-order completion means
      // <= that count implies the *current* tile's loads have landed in LDS.
      __builtin_amdgcn_s_wait_asynccnt(2 * WM + 4);
    } else {
      __builtin_amdgcn_s_wait_asynccnt(0);
    }
#else
    {
      // synchronous fallback: batch all global loads into registers first so
      // the compiler can issue them back-to-back with a single wait.
      v8bf ta[2 * WM], tb[4];
#pragma unroll
      for (int u = 0; u < 2 * WM; ++u) {
        int i = tid + u * 256;
        int r = i >> 3, cc = i & 7;
        ta[u] = *(const v8bf*)&Ab[(long long)r * lda + k0 + cc * 8];
      }
#pragma unroll
      for (int u = 0; u < 4; ++u) {
        int i = tid + u * 256;
        int r = i >> 3, cc = i & 7;
        tb[u] = *(const v8bf*)&Wb[(long long)r * ldw + k0 + cc * 8];
      }
#pragma unroll
      for (int u = 0; u < 2 * WM; ++u) {
        int i = tid + u * 256;
        int r = i >> 3, cc = i & 7;
        *(v8bf*)&sA[cur][r * 64 + cc * 8] = ta[u];
      }
#pragma unroll
      for (int u = 0; u < 4; ++u) {
        int i = tid + u * 256;
        int r = i >> 3, cc = i & 7;
        *(v8bf*)&sB[cur][r * 64 + cc * 8] = tb[u];
      }
      if (k0 + 64 < K)
        __builtin_prefetch(&Ab[(long long)(tid & (MT - 1)) * lda + k0 + 64], 0, 1);
    }
#endif
    __syncthreads();   // tile `cur` fully resident (all waves)

#pragma unroll
    for (int kk = 0; kk < 64; kk += 32) {
      // A fragment (16x32 bf16): lane -> row m=l16; K map: grp g gets
      // K [16g,16g+16) contiguous  -> two 16B LDS loads
      v16bf af[WM];
#pragma unroll
      for (int w = 0; w < WM; ++w) {
        const bf16_t* arow = &sA[cur][(msub * WM * 16 + w * 16 + l16) * 64 + kk];
        v8bf alo = *(const v8bf*)&arow[16 * lg];
        v8bf ahi = *(const v8bf*)&arow[16 * lg + 8];
#pragma unroll
        for (int e = 0; e < 8; ++e) { af[w][e] = alo[e]; af[w][8 + e] = ahi[e]; }
      }
#pragma unroll
      for (int j = 0; j < 4; ++j) {
        // B fragment (32x16 bf16): lane -> col n=l16, K base kk+16*lg,
        // 16 contiguous bf16 -> one aligned 32B LDS load; reused for all WM
        const bf16_t* brow =
            &sB[cur][(nsub * 64 + j * 16 + l16) * 64 + kk + 16 * lg];
        v16bf bfr = *(const v16bf*)brow;
#pragma unroll
        for (int w = 0; w < WM; ++w)
          acc[w][j] = __builtin_amdgcn_wmma_f32_16x16x32_bf16(
              false, af[w], false, bfr, (short)0, acc[w][j], false, false);
      }
    }
    __syncthreads();   // all waves done reading `cur` before it is re-staged
  }

  // ---- epilogue: D layout lane l -> N = l16, M = e + 8*lg ------------------
  float* Cb = C + bz * bsC;
  const float* Cinb = Cin ? (Cin + bz * bsCin) : nullptr;
  const int nbase = blockIdx.x * 128 + nsub * 64;
#pragma unroll
  for (int w = 0; w < WM; ++w) {
    const int mbase = blockIdx.y * MT + msub * WM * 16 + w * 16;
#pragma unroll
    for (int j = 0; j < 4; ++j) {
      int n = nbase + j * 16 + l16;
      float bval = 0.0f;
      if (bias0) bval += bias0[n];
      if (bias1) bval += bias1[n];
#pragma unroll
      for (int e = 0; e < 8; ++e) {
        int m = mbase + e + 8 * lg;
        float v = acc[w][j][e] + bval;
        if (Cinb) v += Cinb[(long long)m * ldcin + n];
        Cb[(long long)m * ldc + n] = v;
      }
    }
  }
}

// ---------------------------------------------------------------------------
// LSTM cell + 4-block single-head attention + residual + LayerNorm.
// One workgroup (256 threads) per batch element b.
// gates layout per b: [NB][4*BS] with torch gate order i,f,g,o.
// ---------------------------------------------------------------------------
__device__ __forceinline__ float sigmf(float x) {
  return 1.0f / (1.0f + __expf(-x));
}

__global__ void __launch_bounds__(256)
k_cell_mha(const float* __restrict__ gates,   // [B][NB*4*BS]
           float* __restrict__ c_st,          // [B][NHID] in-place
           float* __restrict__ h_f32,         // [B][NHID] (post-LN state)
           bf16_t* __restrict__ h_bf16,       // [B][NHID]
           bf16_t* __restrict__ y_out,        // [T*B][NHID] layer output
           const float* __restrict__ wq, const float* __restrict__ bq,
           const float* __restrict__ wk, const float* __restrict__ bk,
           const float* __restrict__ wv, const float* __restrict__ bv,
           const float* __restrict__ wfc, const float* __restrict__ bfc,
           const float* __restrict__ lng, const float* __restrict__ lnb,
           int t) {
  const int b   = blockIdx.x;
  const int tid = threadIdx.x;

  __shared__ float sh[NB][BS];      // h_lstm (pre-attention)
  __shared__ float sq[NB][DK], sk[NB][DK], sv[NB][DK];
  __shared__ float sscore[NB][NB];
  __shared__ float sctx[NB][DK];
  __shared__ float sout[NB][BS];    // residual output pre-LN
  __shared__ float sstat[NB][2];    // mu, var per block row

  const float* gb = gates + (long long)b * (NB * 4 * BS);

  // ---- LSTM cell: 1024 channels, 4 per thread ----
#pragma unroll
  for (int r = 0; r < 4; ++r) {
    int idx = tid + r * 256;
    int n = idx >> 8, j = idx & 255;
    const float* gn = gb + n * (4 * BS);
    float gi = gn[j], gf = gn[BS + j], gg = gn[2 * BS + j], go = gn[3 * BS + j];
    float cp = c_st[(long long)b * NHID + idx];
    float cn = sigmf(gf) * cp + sigmf(gi) * tanhf(gg);
    c_st[(long long)b * NHID + idx] = cn;
    sh[n][j] = sigmf(go) * tanhf(cn);
  }
  __syncthreads();

  // ---- q,k,v projections: [NB][DK] each; threads 0..191 ----
  if (tid < 3 * NB * DK) {
    int which = tid / (NB * DK);
    int r = tid % (NB * DK);
    int n = r >> 4, d = r & 15;
    const float* wrow = (which == 0 ? wq : which == 1 ? wk : wv) + d * BS;
    float a = (which == 0 ? bq : which == 1 ? bk : bv)[d];
#pragma unroll 8
    for (int j = 0; j < BS; ++j) a += sh[n][j] * wrow[j];
    (which == 0 ? sq : which == 1 ? sk : sv)[n][d] = a;
  }
  __syncthreads();

  // ---- scores [NB][NB], scale 1/sqrt(DK)=0.25 ----
  if (tid < NB * NB) {
    int n = tid >> 2, m = tid & 3;
    float s = 0.0f;
#pragma unroll
    for (int d = 0; d < DK; ++d) s += sq[n][d] * sk[m][d];
    sscore[n][m] = s * 0.25f;
  }
  __syncthreads();

  // ---- softmax per row ----
  if (tid < NB) {
    float mx = sscore[tid][0];
#pragma unroll
    for (int m = 1; m < NB; ++m) mx = fmaxf(mx, sscore[tid][m]);
    float sum = 0.0f;
    float e[NB];
#pragma unroll
    for (int m = 0; m < NB; ++m) { e[m] = __expf(sscore[tid][m] - mx); sum += e[m]; }
    float inv = 1.0f / sum;
#pragma unroll
    for (int m = 0; m < NB; ++m) sscore[tid][m] = e[m] * inv;
  }
  __syncthreads();

  // ---- context = attn @ v ----
  if (tid < NB * DK) {
    int n = tid >> 4, d = tid & 15;
    float a = 0.0f;
#pragma unroll
    for (int m = 0; m < NB; ++m) a += sscore[n][m] * sv[m][d];
    sctx[n][d] = a;
  }
  __syncthreads();

  // ---- fc + residual ----
#pragma unroll
  for (int r = 0; r < 4; ++r) {
    int idx = tid + r * 256;
    int n = idx >> 8, j = idx & 255;
    float a = bfc[j];
#pragma unroll
    for (int d = 0; d < DK; ++d) a += sctx[n][d] * wfc[j * DK + d];
    sout[n][j] = a + sh[n][j];
  }
  __syncthreads();

  // ---- LN stats per block row ----
  if (tid < NB) {
    float s = 0.0f, s2 = 0.0f;
    for (int j = 0; j < BS; ++j) { float v = sout[tid][j]; s += v; s2 += v * v; }
    float mu = s * (1.0f / BS);
    sstat[tid][0] = mu;
    sstat[tid][1] = s2 * (1.0f / BS) - mu * mu;
  }
  __syncthreads();

  // ---- normalize, write h (f32 + bf16) and layer output y ----
#pragma unroll
  for (int r = 0; r < 4; ++r) {
    int idx = tid + r * 256;
    int n = idx >> 8, j = idx & 255;
    float inv = rsqrtf(sstat[n][1] + 1e-5f);
    float v = (sout[n][j] - sstat[n][0]) * inv * lng[j] + lnb[j];
    h_f32[(long long)b * NHID + idx] = v;
    bf16_t vb = (bf16_t)v;
    h_bf16[(long long)b * NHID + idx] = vb;
    y_out[((long long)t * BATCH + b) * NHID + idx] = vb;
  }
}

// ---------------------------------------------------------------------------
// Host-side launcher
// ---------------------------------------------------------------------------
extern "C" void kernel_launch(void* const* d_in, const int* in_sizes, int n_in,
                              void* d_out, int out_size, void* d_ws, size_t ws_size,
                              hipStream_t stream) {
  const int*   tokens = (const int*)  d_in[0];
  const float* h0   = (const float*)d_in[1];
  const float* c0   = (const float*)d_in[2];
  const float* embW = (const float*)d_in[3];
  const float* Wih  = (const float*)d_in[4];
  const float* Whh  = (const float*)d_in[5];
  const float* bih  = (const float*)d_in[6];
  const float* bhh  = (const float*)d_in[7];
  const float* wq   = (const float*)d_in[8];
  const float* bq   = (const float*)d_in[9];
  const float* wk   = (const float*)d_in[10];
  const float* bk   = (const float*)d_in[11];
  const float* wv   = (const float*)d_in[12];
  const float* bv   = (const float*)d_in[13];
  const float* wfc  = (const float*)d_in[14];
  const float* bfc  = (const float*)d_in[15];
  const float* lng  = (const float*)d_in[16];
  const float* lnb  = (const float*)d_in[17];
  const float* decW = (const float*)d_in[18];
  const float* decb = (const float*)d_in[19];
  float* out = (float*)d_out;

  const long long TB  = (long long)T_SEQ * BATCH;         // 4096
  const long long DEC = TB * NTOK;                        // 131,072,000
  const long long BN  = (long long)BATCH * NHID;          // 65,536

  // ---- workspace carve-out (~164 MB) ----
  char* ws = (char*)d_ws;
  size_t off = 0;
  auto carve = [&](size_t bytes) -> void* {
    void* p = ws + off;
    off = (off + bytes + 255) & ~(size_t)255;
    return p;
  };
  float*  Gx    = (float*) carve(TB * (size_t)(NB * 4 * BS) * 4);  // 64 MB
  float*  gates = (float*) carve((size_t)BATCH * NB * 4 * BS * 4); //  1 MB
  bf16_t* xA    = (bf16_t*)carve(TB * (size_t)NINP * 2);           //  8 MB
  bf16_t* xB    = (bf16_t*)carve(TB * (size_t)NINP * 2);           //  8 MB
  float*  hF    = (float*) carve(BN * 4);
  bf16_t* hB    = (bf16_t*)carve(BN * 2);
  float*  cF    = (float*) carve(BN * 4);
  bf16_t* WihB  = (bf16_t*)carve((size_t)NLAYERS * NB * 4 * BS * NINP * 2); // 16 MB
  bf16_t* WhhB  = (bf16_t*)carve((size_t)NLAYERS * NB * 4 * BS * BS * 2);   //  4 MB
  bf16_t* decWB = (bf16_t*)carve((size_t)NTOK * NHID * 2);                  // 62.5 MB

  // ---- embedding gather -> xA (bf16) ----
  {
    long long n = TB * NINP;
    k_embed<<<dim3((unsigned)((n + 255) / 256)), dim3(256), 0, stream>>>(
        tokens, embW, xA, n);
  }
  // ---- one-time weight conversions ----
  {
    long long n = (long long)NLAYERS * NB * 4 * BS * NINP;
    k_f32_to_bf16<<<dim3((unsigned)((n + 255) / 256)), dim3(256), 0, stream>>>(Wih, WihB, n);
    n = (long long)NLAYERS * NB * 4 * BS * BS;
    k_f32_to_bf16<<<dim3((unsigned)((n + 255) / 256)), dim3(256), 0, stream>>>(Whh, WhhB, n);
    n = (long long)NTOK * NHID;
    k_f32_to_bf16<<<dim3((unsigned)((n + 255) / 256)), dim3(256), 0, stream>>>(decW, decWB, n);
  }

  for (int l = 0; l < NLAYERS; ++l) {
    const bf16_t* xin  = (l == 0) ? xA : xB;
    bf16_t*       yout = (l == 0) ? xB : xA;

    // init states from h0/c0
    (void)hipMemcpyAsync(hF, h0 + (long long)l * BN, BN * 4,
                         hipMemcpyDeviceToDevice, stream);
    (void)hipMemcpyAsync(cF, c0 + (long long)l * BN, BN * 4,
                         hipMemcpyDeviceToDevice, stream);
    k_f32_to_bf16<<<dim3((unsigned)((BN + 255) / 256)), dim3(256), 0, stream>>>(
        h0 + (long long)l * BN, hB, BN);

    // Gx[T*B, 4096] = X * Wih_l^T + b_ih_l + b_hh_l   (big WMMA GEMM, 128x128)
    k_gemm_bf16<2><<<dim3(32, 32, 1), dim3(256), 0, stream>>>(
        xin, NINP, 0,
        WihB + (long long)l * NB * 4 * BS * NINP, NINP, 0,
        Gx, NB * 4 * BS, 0,
        nullptr, 0, 0,
        bih + (long long)l * NB * 4 * BS, bhh + (long long)l * NB * 4 * BS,
        NINP);

    for (int t = 0; t < T_SEQ; ++t) {
      // gates[B,4096] = Gx[t] + blockdiag_n( h[:, n*256:] * Whh_l[n]^T )
      k_gemm_bf16<1><<<dim3(8, 1, NB), dim3(256), 0, stream>>>(
          hB, NHID, BS,                                    // A: h, batch stride 256 cols
          WhhB + (long long)l * NB * 4 * BS * BS, BS,
          (long long)(4 * BS) * BS,                        // W per block
          gates, NB * 4 * BS, 4 * BS,                      // C slice per block
          Gx + (long long)t * BATCH * (NB * 4 * BS), NB * 4 * BS, 4 * BS,
          nullptr, nullptr,
          BS);
      // LSTM cell + MHA + LN; updates h/c, emits y[t]
      k_cell_mha<<<dim3(BATCH), dim3(256), 0, stream>>>(
          gates, cF, hF, hB, yout,
          wq, bq, wk, bk, wv, bv, wfc, bfc, lng, lnb, t);
    }

    // final states -> d_out tail: new_h at DEC + l*BN, new_c at DEC + 2*BN + l*BN
    (void)hipMemcpyAsync(out + DEC + (long long)l * BN, hF, BN * 4,
                         hipMemcpyDeviceToDevice, stream);
    (void)hipMemcpyAsync(out + DEC + 2 * BN + (long long)l * BN, cF, BN * 4,
                         hipMemcpyDeviceToDevice, stream);
  }

  // ---- decoder: decoded[T*B, NTOK] = Y * decW^T + dec_b  (268 GFLOP) ----
  {
    const bf16_t* Y = (NLAYERS % 2 == 0) ? xA : xB;  // last layer wrote here
    k_gemm_bf16<2><<<dim3(NTOK / 128, (unsigned)(TB / 128), 1), dim3(256), 0, stream>>>(
        Y, NHID, 0,
        decWB, NHID, 0,
        out, NTOK, 0,
        nullptr, 0, 0,
        decb, nullptr,
        NHID);
  }
}
